// VQCodebook_18889266168138
// MI455X (gfx1250) — matrix operand order repaired and checked by reading, not compiled, not run on hardware
//
#include <hip/hip_runtime.h>
#include <stdint.h>

#define NUM_CODE 8192
#define CODE_DIM 256
#define NROWS    16384   // 16*32*32 query rows
#define MT       1024    // NROWS/16
#define DTILES   8       // CODE_DIM/32
#define KTILES   512     // NUM_CODE/16

typedef __attribute__((ext_vector_type(16))) __bf16 v16bf;
typedef __attribute__((ext_vector_type(8)))  float  v8f;

union Frag  { uint32_t u[8]; uint4 q[2]; v16bf bf; };
union FragF { float f[8]; v8f v; };

__device__ inline uint16_t f32_to_bf16(float f) {
  uint32_t u = __float_as_uint(f);
  u += 0x7FFFu + ((u >> 16) & 1u);          // round-to-nearest-even
  return (uint16_t)(u >> 16);
}
__device__ inline float bf16_as_f32(uint16_t h) {
  return __uint_as_float(((uint32_t)h) << 16);
}

// ---------------------------------------------------------------------------
// Pack z_e (NCHW fp32) into WMMA A-fragment order, bf16 hi/lo split.
// A layout (16-bit A 16x32, wave32): lane = half*16 + m,
//   VGPR v holds K = {k0, k0+1} with k0 = 2v + (v>=4 ? 8 : 0) + 8*half.
// Flat index: ((mt*8 + dt)*32 + lane)*8 + v  == tid decomposition below.
// ---------------------------------------------------------------------------
__global__ void vq_pack_a(const float* __restrict__ z,
                          uint32_t* __restrict__ Ah, uint32_t* __restrict__ Al) {
  uint32_t tid  = blockIdx.x * 256u + threadIdx.x;   // 2,097,152 threads
  uint32_t v    = tid & 7u;
  uint32_t lane = (tid >> 3) & 31u;
  uint32_t dt   = (tid >> 8) & 7u;
  uint32_t mt   = tid >> 11;
  uint32_t m    = lane & 15u, half = lane >> 4;
  uint32_t n    = mt * 16u + m;
  uint32_t b    = n >> 10, hw = n & 1023u;           // H*W = 1024
  uint32_t k0   = 2u * v + ((v >= 4u) ? 8u : 0u) + 8u * half;
  uint32_t d    = dt * 32u + k0;
  size_t   base = (size_t)b * (CODE_DIM * 1024) + hw;
  float s0 = z[base + (size_t)d       * 1024];
  float s1 = z[base + (size_t)(d + 1) * 1024];
  uint16_t h0 = f32_to_bf16(s0), h1 = f32_to_bf16(s1);
  uint16_t l0 = f32_to_bf16(s0 - bf16_as_f32(h0));
  uint16_t l1 = f32_to_bf16(s1 - bf16_as_f32(h1));
  Ah[tid] = ((uint32_t)h1 << 16) | h0;
  Al[tid] = ((uint32_t)l1 << 16) | l0;
}

// ---------------------------------------------------------------------------
// Pack codebook^T into WMMA B-fragment order (32x16 bf16), hi/lo split.
// B layout (wave32): lane = half*16 + col; VGPR v holds Kdim = {dd, dd+1}
// with dd = 2v + (v>=4 ? 8 : 0) + 8*half.  Flat index: ((dt*KTILES + kt)*32 + lane)*8 + v.
// ---------------------------------------------------------------------------
__global__ void vq_pack_b(const float* __restrict__ cb,
                          uint32_t* __restrict__ Bh, uint32_t* __restrict__ Bl) {
  uint32_t tid  = blockIdx.x * 256u + threadIdx.x;   // 1,048,576 threads
  uint32_t v    = tid & 7u;
  uint32_t lane = (tid >> 3) & 31u;
  uint32_t kt   = (tid >> 8) & 511u;
  uint32_t dt   = tid >> 17;
  uint32_t col  = lane & 15u, half = lane >> 4;
  uint32_t dd   = 2u * v + ((v >= 4u) ? 8u : 0u) + 8u * half;
  uint32_t code = kt * 16u + col;
  uint32_t d    = dt * 32u + dd;
  const float* src = cb + (size_t)code * CODE_DIM + d;   // contiguous pair
  float s0 = src[0], s1 = src[1];
  uint16_t h0 = f32_to_bf16(s0), h1 = f32_to_bf16(s1);
  uint16_t l0 = f32_to_bf16(s0 - bf16_as_f32(h0));
  uint16_t l1 = f32_to_bf16(s1 - bf16_as_f32(h1));
  Bh[tid] = ((uint32_t)h1 << 16) | h0;
  Bl[tid] = ((uint32_t)l1 << 16) | l0;
}

// ---------------------------------------------------------------------------
// c2[k] = sum_d cb[k,d]^2  (fp32, wave per code)
// ---------------------------------------------------------------------------
__global__ void vq_c2(const float* __restrict__ cb, float* __restrict__ c2) {
  uint32_t wave = threadIdx.x >> 5, lane = threadIdx.x & 31u;
  uint32_t code = blockIdx.x * 8u + wave;
  const float* row = cb + (size_t)code * CODE_DIM;
  float s = 0.f;
#pragma unroll
  for (int i = 0; i < 8; ++i) { float x = row[lane + 32 * i]; s += x * x; }
#pragma unroll
  for (int off = 16; off >= 1; off >>= 1) s += __shfl_xor(s, off, 32);
  if (lane == 0) c2[code] = s;
}

// ---------------------------------------------------------------------------
// Main: per block, one 16-row M tile; 8 waves sweep all K tiles with bf16x3
// split WMMA; in-register argmin; gather fp32 codebook rows to NCHW output.
// ---------------------------------------------------------------------------
__global__ void __launch_bounds__(256)
vq_main(const uint32_t* __restrict__ Ah, const uint32_t* __restrict__ Al,
        const uint32_t* __restrict__ Bh, const uint32_t* __restrict__ Bl,
        const float* __restrict__ c2, const float* __restrict__ cb,
        float* __restrict__ out_zq, float* __restrict__ out_idx) {
  __shared__ float sVal[8][16];
  __shared__ int   sIdx[8][16];

  const uint32_t mt = blockIdx.x;
  const uint32_t wave = threadIdx.x >> 5, lane = threadIdx.x & 31u;

  // A fragments for this M tile stay resident in VGPRs (hi + lo).
  Frag ah[DTILES], al[DTILES];
#pragma unroll
  for (int dt = 0; dt < DTILES; ++dt) {
    const uint4* ph = (const uint4*)(Ah + ((size_t)(mt * 8u + dt) * 32u + lane) * 8u);
    ah[dt].q[0] = ph[0]; ah[dt].q[1] = ph[1];
    const uint4* pl = (const uint4*)(Al + ((size_t)(mt * 8u + dt) * 32u + lane) * 8u);
    al[dt].q[0] = pl[0]; al[dt].q[1] = pl[1];
  }

  float bestv[8]; int besti[8];
#pragma unroll
  for (int i = 0; i < 8; ++i) { bestv[i] = 3.4e38f; besti[i] = 0; }

  for (uint32_t kt = wave; kt < KTILES; kt += 8u) {   // wave-uniform loop
    FragF acc;
#pragma unroll
    for (int i = 0; i < 8; ++i) acc.f[i] = 0.f;
#pragma unroll
    for (int dt = 0; dt < DTILES; ++dt) {
      Frag bh, bl;
      const uint4* pb = (const uint4*)(Bh + ((size_t)(dt * KTILES + kt) * 32u + lane) * 8u);
      bh.q[0] = pb[0]; bh.q[1] = pb[1];
      const uint4* pl = (const uint4*)(Bl + ((size_t)(dt * KTILES + kt) * 32u + lane) * 8u);
      bl.q[0] = pl[0]; bl.q[1] = pl[1];
      // z.c ~= zl*ch + zh*cl + zh*ch (chained accumulation)
      acc.v = __builtin_amdgcn_wmma_f32_16x16x32_bf16(false, al[dt].bf, false, bh.bf,
                                                      (short)0, acc.v, false, false);
      acc.v = __builtin_amdgcn_wmma_f32_16x16x32_bf16(false, ah[dt].bf, false, bl.bf,
                                                      (short)0, acc.v, false, false);
      acc.v = __builtin_amdgcn_wmma_f32_16x16x32_bf16(false, ah[dt].bf, false, bh.bf,
                                                      (short)0, acc.v, false, false);
    }
    // C layout: lane<16 -> N=lane, M=v ; lane>=16 -> N=lane-16, M=8+v
    int code = (int)(kt * 16u + (lane & 15u));
    float cc = c2[code];
#pragma unroll
    for (int v2 = 0; v2 < 8; ++v2) {
      float sc = cc - 2.f * acc.f[v2];            // z2 term is row-constant
      if (sc < bestv[v2]) { bestv[v2] = sc; besti[v2] = code; }
    }
  }

  // Reduce argmin across the 16 lanes sharing each row (xor within halves).
#pragma unroll
  for (int v2 = 0; v2 < 8; ++v2) {
    float bv = bestv[v2]; int bi = besti[v2];
#pragma unroll
    for (int off = 1; off < 16; off <<= 1) {
      float ov = __shfl_xor(bv, off, 32);
      int   oi = __shfl_xor(bi, off, 32);
      if (ov < bv || (ov == bv && oi < bi)) { bv = ov; bi = oi; }
    }
    if ((lane & 15u) == 0u) {
      int row = (int)(lane >> 4) * 8 + v2;
      sVal[wave][row] = bv; sIdx[wave][row] = bi;
    }
  }
  __syncthreads();

  // Reduce across the 8 waves; emit indices.
  if (threadIdx.x < 16) {
    int row = (int)threadIdx.x;
    float bv = sVal[0][row]; int bi = sIdx[0][row];
#pragma unroll
    for (int w = 1; w < 8; ++w) {
      float ov = sVal[w][row]; int oi = sIdx[w][row];
      if (ov < bv || (ov == bv && oi < bi)) { bv = ov; bi = oi; }
    }
    sIdx[0][row] = bi;
    out_idx[mt * 16 + row] = (float)bi;
  }
  __syncthreads();

  // Gather winning fp32 codebook rows to NCHW output.
  {
    uint32_t dcol = threadIdx.x;                  // 0..255 == CODE_DIM
    uint32_t n0 = mt * 16u;
    uint32_t b = n0 >> 10, hw0 = n0 & 1023u;      // block never crosses batch
#pragma unroll
    for (int row = 0; row < 16; ++row) {
      int bi = sIdx[0][row];
      out_zq[(size_t)b * (CODE_DIM * 1024) + (size_t)dcol * 1024u + hw0 + row] =
          cb[(size_t)bi * CODE_DIM + dcol];
    }
  }
}

// ---------------------------------------------------------------------------
extern "C" void kernel_launch(void* const* d_in, const int* in_sizes, int n_in,
                              void* d_out, int out_size, void* d_ws, size_t ws_size,
                              hipStream_t stream) {
  const float* z_e = (const float*)d_in[0];   // [16,256,32,32] fp32
  const float* cb  = (const float*)d_in[1];   // [8192,256] fp32
  float* out = (float*)d_out;                 // z_q (4,194,304) ++ idx (16,384)

  uint8_t* ws = (uint8_t*)d_ws;
  uint32_t* Ah = (uint32_t*)(ws);                               // 8 MB
  uint32_t* Al = (uint32_t*)(ws + (size_t)8  * 1024 * 1024);    // 8 MB
  uint32_t* Bh = (uint32_t*)(ws + (size_t)16 * 1024 * 1024);    // 4 MB
  uint32_t* Bl = (uint32_t*)(ws + (size_t)20 * 1024 * 1024);    // 4 MB
  float*    c2 = (float*)   (ws + (size_t)24 * 1024 * 1024);    // 32 KB

  vq_pack_a<<<8192, 256, 0, stream>>>(z_e, Ah, Al);
  vq_pack_b<<<4096, 256, 0, stream>>>(cb, Bh, Bl);
  vq_c2    <<<1024, 256, 0, stream>>>(cb, c2);
  vq_main  <<<MT,   256, 0, stream>>>(Ah, Al, Bh, Bl, c2, cb,
                                      out, out + (size_t)NROWS * CODE_DIM);
}
